// ModelWithLoss_67808943669608
// MI455X (gfx1250) — matrix operand order repaired and checked by reading, not compiled, not run on hardware
//
#include <hip/hip_runtime.h>

typedef __bf16 bf16_t;
typedef __attribute__((ext_vector_type(16))) __bf16 v16bf;
typedef __attribute__((ext_vector_type(8)))  __bf16 v8bf;
typedef __attribute__((ext_vector_type(8)))  float  v8f;
typedef __attribute__((ext_vector_type(4)))  unsigned int v4u;
typedef __attribute__((ext_vector_type(8)))  int    v8i;
typedef __attribute__((ext_vector_type(4)))  int    v4i;
typedef unsigned int u32;

#define B_SZ 512
#define D_SZ 2048
#define C_SZ 7178
#define P_SZ 8
#define C_PAD 7232   // C rounded up to 64

#define BM 128
#define BN 64
#define BK 32
#define XS 40        // LDS row stride in elements: 64B tile row + 16B TDM pad

// ---------------------------------------------------------------------------
// TDM 2D tile load: build D# groups per CDNA5 ISA ch.8 and issue
// tensor_load_to_lds. data_size = 2 bytes (bf16). LDS padding: +16B per 64B.
// ---------------------------------------------------------------------------
__device__ __forceinline__ void tdm_load_tile(u32 lds_addr, const bf16_t* gptr,
                                              u32 tdim0, u32 tdim1,
                                              u32 tile0, u32 tile1, u32 stride0) {
  unsigned long long ga = (unsigned long long)(size_t)gptr;
  v4u g0;
  g0[0] = 1u;                                         // count=1 (valid), no gather
  g0[1] = lds_addr;                                   // LDS byte address
  g0[2] = (u32)(ga & 0xFFFFFFFFu);                    // global_addr[31:0]
  g0[3] = (u32)((ga >> 32) & 0x01FFFFFFu) | (2u << 30); // addr[56:32] | type=2
  v8i g1;
  g1[0] = (int)((1u << 16)      // data_size = 1 -> 2 bytes
              | (1u << 20)      // pad_enable
              | (3u << 22)      // pad_interval: after every 64 bytes
              | (3u << 25));    // pad_amount: 4 DWORDs = 16 bytes
  g1[1] = (int)((tdim0 & 0xFFFFu) << 16);                       // tensor_dim0 lo
  g1[2] = (int)((tdim0 >> 16) | ((tdim1 & 0xFFFFu) << 16));     // dim0 hi|dim1 lo
  g1[3] = (int)((tdim1 >> 16) | (tile0 << 16));                 // dim1 hi|tile0
  g1[4] = (int)tile1;                                           // tile1, tile2=0
  g1[5] = (int)stride0;                                         // dim0_stride lo
  g1[6] = 0;                                                    // stride hi bits
  g1[7] = 0;
  v4i z = {0, 0, 0, 0};
#if defined(__clang_major__) && __clang_major__ >= 23
  v8i z8 = {0, 0, 0, 0, 0, 0, 0, 0};
  __builtin_amdgcn_tensor_load_to_lds(g0, g1, z, z, z8, 0);
#else
  __builtin_amdgcn_tensor_load_to_lds(g0, g1, z, z, 0);
#endif
}

// ---------------------------------------------------------------------------
// Kernel 0: split fp32 matrix into hi/lo bf16 (x ~= hi + lo); zero pad rows.
// ---------------------------------------------------------------------------
__global__ __launch_bounds__(256)
void split_mat(const float* __restrict__ src, bf16_t* __restrict__ hi,
               bf16_t* __restrict__ lo, long long n_src, long long n_total) {
  long long i = ((long long)blockIdx.x * 256 + threadIdx.x) * 4;
  if (i >= n_total) return;
  float f[4];
  if (i + 4 <= n_src) {
    float4 v = *(const float4*)(src + i);
    f[0] = v.x; f[1] = v.y; f[2] = v.z; f[3] = v.w;
  } else {
    #pragma unroll
    for (int e = 0; e < 4; ++e) f[e] = (i + e < n_src) ? src[i + e] : 0.0f;
  }
  #pragma unroll
  for (int e = 0; e < 4; ++e) {
    bf16_t h = (bf16_t)f[e];
    hi[i + e] = h;
    lo[i + e] = (bf16_t)(f[e] - (float)h);
  }
}

// ---------------------------------------------------------------------------
// Kernel 1: logits = x @ W^T via split-bf16 WMMA; TDM double-buffered staging.
// ---------------------------------------------------------------------------
__global__ __launch_bounds__(256)
void gemm_logits(const bf16_t* __restrict__ Xh, const bf16_t* __restrict__ Xl,
                 const bf16_t* __restrict__ Wh, const bf16_t* __restrict__ Wl,
                 float* __restrict__ logits) {
  __shared__ bf16_t sxh[2][BM][XS];
  __shared__ bf16_t sxl[2][BM][XS];
  __shared__ bf16_t swh[2][BN][XS];
  __shared__ bf16_t swl[2][BN][XS];

  const int tid  = threadIdx.x;
  const int wave = tid >> 5;
  const int lane = tid & 31;
  const int N0 = blockIdx.x * BN;
  const int M0 = blockIdx.y * BM;

  v8f acc[4];
  #pragma unroll
  for (int s = 0; s < 4; ++s)
    #pragma unroll
    for (int e = 0; e < 8; ++e) acc[s][e] = 0.0f;

  const int lhalf = lane >> 4;
  const int lmod  = lane & 15;
  const int mrow  = wave * 16 + lmod;
  const int a_k0  = lhalf * 8;
  const int b_k0  = lhalf * 8;

  const size_t xbase = (size_t)M0 * D_SZ;
  const size_t wbase = (size_t)N0 * D_SZ;
  const int NCH = D_SZ / BK;   // 64 K-chunks

  // prologue: chunk 0 -> buffer 0 (wave 0 issues all 4 TDM tile loads)
  if (wave == 0) {
    tdm_load_tile((u32)(size_t)&sxh[0][0][0], Xh + xbase, D_SZ, B_SZ,  BK, BM, D_SZ);
    tdm_load_tile((u32)(size_t)&sxl[0][0][0], Xl + xbase, D_SZ, B_SZ,  BK, BM, D_SZ);
    tdm_load_tile((u32)(size_t)&swh[0][0][0], Wh + wbase, D_SZ, C_PAD, BK, BN, D_SZ);
    tdm_load_tile((u32)(size_t)&swl[0][0][0], Wl + wbase, D_SZ, C_PAD, BK, BN, D_SZ);
  }

  for (int ch = 0; ch < NCH; ++ch) {
    const int cb = ch & 1;
    if (wave == 0) {
      if (ch + 1 < NCH) {
        const int nb = (ch + 1) & 1;
        const int kn = (ch + 1) * BK;
        tdm_load_tile((u32)(size_t)&sxh[nb][0][0], Xh + xbase + kn, D_SZ, B_SZ,  BK, BM, D_SZ);
        tdm_load_tile((u32)(size_t)&sxl[nb][0][0], Xl + xbase + kn, D_SZ, B_SZ,  BK, BM, D_SZ);
        tdm_load_tile((u32)(size_t)&swh[nb][0][0], Wh + wbase + kn, D_SZ, C_PAD, BK, BN, D_SZ);
        tdm_load_tile((u32)(size_t)&swl[nb][0][0], Wl + wbase + kn, D_SZ, C_PAD, BK, BN, D_SZ);
        __builtin_amdgcn_s_wait_tensorcnt(4);   // chunk ch done; ch+1 in flight
      } else {
        __builtin_amdgcn_s_wait_tensorcnt(0);
      }
    }
    __syncthreads();   // chunk ch visible to all waves

    // A fragments (16x32 bf16 layout, ISA 7.12.2)
    union { v16bf v; v8bf h[2]; } ah, al;
    ah.h[0] = *(const v8bf*)&sxh[cb][mrow][a_k0];
    ah.h[1] = *(const v8bf*)&sxh[cb][mrow][16 + a_k0];
    al.h[0] = *(const v8bf*)&sxl[cb][mrow][a_k0];
    al.h[1] = *(const v8bf*)&sxl[cb][mrow][16 + a_k0];

    #pragma unroll
    for (int st = 0; st < 4; ++st) {
      const int n = st * 16 + lmod;
      union { v16bf v; v8bf h[2]; } bh, bl;
      bh.h[0] = *(const v8bf*)&swh[cb][n][b_k0];
      bh.h[1] = *(const v8bf*)&swh[cb][n][b_k0 + 16];
      bl.h[0] = *(const v8bf*)&swl[cb][n][b_k0];
      bl.h[1] = *(const v8bf*)&swl[cb][n][b_k0 + 16];
      acc[st] = __builtin_amdgcn_wmma_f32_16x16x32_bf16(
          false, ah.v, false, bh.v, (short)0, acc[st], false, false);
      acc[st] = __builtin_amdgcn_wmma_f32_16x16x32_bf16(
          false, ah.v, false, bl.v, (short)0, acc[st], false, false);
      acc[st] = __builtin_amdgcn_wmma_f32_16x16x32_bf16(
          false, al.v, false, bh.v, (short)0, acc[st], false, false);
    }
    __syncthreads();   // all reads of buffer cb done before TDM overwrites it
  }

  // epilogue: C/D layout (VGPR r: M = r + lhalf*8, N = lane%16)
  #pragma unroll
  for (int st = 0; st < 4; ++st) {
    const int n = N0 + st * 16 + lmod;
    if (n < C_SZ) {
      const int mbase = M0 + wave * 16 + lhalf * 8;
      #pragma unroll
      for (int r = 0; r < 8; ++r)
        logits[(size_t)(mbase + r) * C_SZ + n] = acc[st][r];
    }
  }
}

// ---------------------------------------------------------------------------
// Kernel 2: per-sample online LSE over negatives + fold in each positive
// ---------------------------------------------------------------------------
__global__ __launch_bounds__(256)
void lse_loss(const float* __restrict__ logits, const float* __restrict__ bias,
              const int* __restrict__ labels, float* __restrict__ partials) {
  const int b   = blockIdx.x;
  const int tid = threadIdx.x;

  int lab[P_SZ];
  #pragma unroll
  for (int j = 0; j < P_SZ; ++j) lab[j] = labels[b * P_SZ + j];

  float m = -3.4e38f, s = 0.0f;
  for (int c = tid; c < C_SZ; c += 256) {
    bool skip = false;
    #pragma unroll
    for (int j = 0; j < P_SZ; ++j) skip |= (c == lab[j]);
    if (skip) continue;
    float v = logits[(size_t)b * C_SZ + c] + bias[c];
    if (v > m) { s = s * expf(m - v) + 1.0f; m = v; }
    else       { s += expf(v - m); }
  }

  __shared__ float sm[256];
  __shared__ float ss[256];
  sm[tid] = m; ss[tid] = s;
  __syncthreads();
  for (int off = 128; off > 0; off >>= 1) {
    if (tid < off) {
      float m2 = sm[tid + off], s2 = ss[tid + off];
      float M  = fmaxf(sm[tid], m2);
      ss[tid]  = ss[tid] * expf(sm[tid] - M) + s2 * expf(m2 - M);
      sm[tid]  = M;
    }
    __syncthreads();
  }

  if (tid == 0) {
    const float Mn = sm[0], Sn = ss[0];
    float total = 0.0f;
    #pragma unroll
    for (int j = 0; j < P_SZ; ++j) {
      float lp  = logits[(size_t)b * C_SZ + lab[j]] + bias[lab[j]];
      float M2  = fmaxf(Mn, lp);
      float lse = M2 + logf(Sn * expf(Mn - M2) + expf(lp - M2));
      total += lse - lp;
    }
    partials[b] = total;
  }
}

// ---------------------------------------------------------------------------
// Kernel 3: deterministic fixed-order final reduction -> mean over B*P
// ---------------------------------------------------------------------------
__global__ __launch_bounds__(256)
void reduce_loss(const float* __restrict__ partials, float* __restrict__ out) {
  __shared__ float sh[256];
  const int tid = threadIdx.x;
  sh[tid] = partials[tid] + partials[tid + 256];
  __syncthreads();
  for (int off = 128; off > 0; off >>= 1) {
    if (tid < off) sh[tid] += sh[tid + off];
    __syncthreads();
  }
  if (tid == 0) out[0] = sh[0] / (float)(B_SZ * P_SZ);
}

// ---------------------------------------------------------------------------
extern "C" void kernel_launch(void* const* d_in, const int* in_sizes, int n_in,
                              void* d_out, int out_size, void* d_ws, size_t ws_size,
                              hipStream_t stream) {
  const float* X      = (const float*)d_in[0];   // [B, D]
  const float* W      = (const float*)d_in[1];   // [C, D]
  const float* bias   = (const float*)d_in[2];   // [C]
  const int*   labels = (const int*)d_in[3];     // [B, P]

  // workspace layout (bytes)
  char* ws = (char*)d_ws;
  float*  logits = (float*)ws;                                   // B*C f32
  size_t  off    = (size_t)B_SZ * C_SZ * sizeof(float);
  bf16_t* Xh = (bf16_t*)(ws + off);  off += (size_t)B_SZ * D_SZ * 2;
  bf16_t* Xl = (bf16_t*)(ws + off);  off += (size_t)B_SZ * D_SZ * 2;
  bf16_t* Wh = (bf16_t*)(ws + off);  off += (size_t)C_PAD * D_SZ * 2;
  bf16_t* Wl = (bf16_t*)(ws + off);  off += (size_t)C_PAD * D_SZ * 2;
  float*  partials = (float*)(ws + off);

  const long long NX  = (long long)B_SZ * D_SZ;    // x elements
  const long long NWS = (long long)C_SZ * D_SZ;    // W source elements
  const long long NWT = (long long)C_PAD * D_SZ;   // W padded elements

  split_mat<<<(unsigned)((NX / 4 + 255) / 256), 256, 0, stream>>>(X, Xh, Xl, NX, NX);
  split_mat<<<(unsigned)((NWT / 4 + 255) / 256), 256, 0, stream>>>(W, Wh, Wl, NWS, NWT);

  dim3 g1(C_PAD / BN, B_SZ / BM);    // 113 x 4
  gemm_logits<<<g1, 256, 0, stream>>>(Xh, Xl, Wh, Wl, logits);
  lse_loss<<<B_SZ, 256, 0, stream>>>(logits, bias, labels, partials);
  reduce_loss<<<1, 256, 0, stream>>>(partials, (float*)d_out);
}
